// TransparentRNN_63256278335864
// MI455X (gfx1250) — matrix operand order repaired and checked by reading, not compiled, not run on hardware
//
#include <hip/hip_runtime.h>
#include <hip/hip_bf16.h>
#include <stdint.h>

// Problem constants
#define B_    256
#define T_    512
#define DIN_  256
#define H_    512
#define DOUT_ 64
#define L_    2
#define EPS_  1e-5f
#define HK_   1024            // 2*H (GEMM K)
#define GROUPS_ 16            // B/16 batch blocks
#define PARTS_  4             // column split per group
#define PCOLS_  128           // H/PARTS

typedef __attribute__((ext_vector_type(16))) __bf16 v16bf;
typedef __attribute__((ext_vector_type(8)))  float  v8f;
typedef int v4i_ __attribute__((vector_size(16)));

union Frag {
  v16bf bf;
  uint4 q[2];
  unsigned short u[16];
};

__device__ __forceinline__ unsigned short f2bf(float f) {
  union { float f; unsigned u; } c; c.f = f;
  unsigned r = c.u + 0x7FFFu + ((c.u >> 16) & 1u);
  return (unsigned short)(r >> 16);
}
__device__ __forceinline__ float bf2f(unsigned short u) {
  union { unsigned u; float f; } c; c.u = ((unsigned)u) << 16;
  return c.f;
}

// ---------------- CDNA5 async global->LDS helpers ----------------
#if defined(__has_builtin)
#if __has_builtin(__builtin_amdgcn_global_load_async_to_lds_b128)
#define USE_ASYNC_BUILTIN 1
#endif
#if __has_builtin(__builtin_amdgcn_s_wait_asynccnt)
#define USE_WAITASYNC_BUILTIN 1
#endif
#endif

__device__ __forceinline__ void async_ld_b128(const unsigned short* g,
                                              unsigned short* l,
                                              unsigned ldsOff) {
#ifdef USE_ASYNC_BUILTIN
  (void)ldsOff;
  __builtin_amdgcn_global_load_async_to_lds_b128(
      (__attribute__((address_space(1))) v4i_*)(void*)g,
      (__attribute__((address_space(3))) v4i_*)(void*)l, 0, 0);
#else
  (void)l;
  asm volatile("global_load_async_to_lds_b128 %0, %1, off"
               :: "v"(ldsOff), "v"(g) : "memory");
#endif
}

__device__ __forceinline__ void wait_asynccnt0() {
#ifdef USE_WAITASYNC_BUILTIN
  __builtin_amdgcn_s_wait_asynccnt(0);
#else
  asm volatile("s_wait_asynccnt 0x0" ::: "memory");
#endif
}

// ---------------- workspace layout (bytes) ----------------
#define XPK_OFF   0ull
#define XPK_BYTES (512ull*16*16*512*2)       // packed bf16 x@Wi+bi A-fragments: [t][g][kt16][lane32][slot16]
#define WPK_OFF   (XPK_OFF + XPK_BYTES)
#define WPK_BYTES (2ull*3*32*32*512*2)       // packed bf16 Wz/Wr/Wc B-fragments
#define WIPK_OFF  (WPK_OFF + WPK_BYTES)
#define WIPK_BYTES (8ull*32*512*2)           // packed bf16 Wi B-fragments
#define XRH_OFF   (WIPK_OFF + WIPK_BYTES)
#define XRH_BYTES (16ull*4*16*128*4)         // r*h exchange f32 [g][part][row16][col128]
#define XH_OFF    (XRH_OFF + XRH_BYTES)
#define XH_BYTES  (16ull*4*16*128*4)         // h_new exchange f32
#define XST_OFF   (XH_OFF + XH_BYTES)
#define XST_BYTES (16ull*4*16*2*4)           // per-part row (sum,sumsq)
#define CNT_OFF   (XST_OFF + XST_BYTES)      // 16 spin-barrier counters

// ---------------- scan LDS layout (bytes within dynamic smem) ----------------
#define S_XSTAGE  0        // 2 x 8192 ushort (double-buffered packed x tile)
#define S_HPK0    32768    // 8192 ushort
#define S_HPK1    49152    // 8192 ushort
#define S_RHPK    65536    // 8192 ushort
#define S_HSTATE  81920    // 4096 ushort  [2][16][128] bf16
#define S_HNB     90112    // 2048 ushort  [16][128] bf16
#define S_GB      94208    // 2048 f32     [l][{gamma,beta}][512]
#define S_PSUM    102400   // 256 f32
#define S_PSQ     103424   // 256 f32
#define S_MU      104448   // 16 f32
#define S_RSTD    104512   // 16 f32
#define S_TOTAL   104576

// ---------------- group spin barrier (4 WGs per group) ----------------
__device__ __forceinline__ void group_barrier(unsigned* cnt, unsigned target4) {
  __threadfence();
  __syncthreads();
  if (threadIdx.x == 0) {
    __hip_atomic_fetch_add(cnt, 1u, __ATOMIC_RELEASE, __HIP_MEMORY_SCOPE_AGENT);
    unsigned v;
    do {
      v = __hip_atomic_load(cnt, __ATOMIC_ACQUIRE, __HIP_MEMORY_SCOPE_AGENT);
      if (v < target4) __builtin_amdgcn_s_sleep(1);
    } while (v < target4);
  }
  __syncthreads();
}

// ---------------- tiny init kernel ----------------
__global__ void k_zero(unsigned* cnt) {
  if (threadIdx.x < GROUPS_) cnt[threadIdx.x] = 0u;
}

// ---------------- weight packing: f32 -> bf16 WMMA B-fragment order ----------------
// B-frag (16x16x32 bf16): N=lane&15, K=kt*32+8*(lane>>4)+(slot&7)+16*(slot>>3)
__global__ void k_pack_w(const float* __restrict__ Wz, const float* __restrict__ Wr,
                         const float* __restrict__ Wc, unsigned short* __restrict__ wpk) {
  int idx = blockIdx.x * 256 + threadIdx.x;          // 3,145,728 total
  int s    = idx & 15;
  int lane = (idx >> 4) & 31;
  int nt   = (idx >> 9) & 31;
  int kt   = (idx >> 14) & 31;
  int gg   = idx >> 19;                              // 0..5
  int gate = gg % 3, l = gg / 3;
  int n = nt * 16 + (lane & 15);
  int k = kt * 32 + ((lane >> 4) << 3) + (s & 7) + ((s >> 3) << 4);
  const float* W = (gate == 0) ? Wz : ((gate == 1) ? Wr : Wc);
  wpk[idx] = f2bf(W[(size_t)(l * HK_ + k) * H_ + n]);
}

__global__ void k_pack_wi(const float* __restrict__ Wi, unsigned short* __restrict__ wipk) {
  int idx = blockIdx.x * 256 + threadIdx.x;          // 131,072 total
  int s    = idx & 15;
  int lane = (idx >> 4) & 31;
  int nt   = (idx >> 9) & 31;
  int kt   = (idx >> 14) & 7;
  int n = nt * 16 + (lane & 15);
  int k = kt * 32 + ((lane >> 4) << 3) + (s & 7) + ((s >> 3) << 4);
  wipk[idx] = f2bf(Wi[(size_t)k * H_ + n]);
}

// ---------------- input projection: x@Wi+bi -> packed bf16 A-fragments ----------------
// A-frag (16x16x32 bf16): lane = m + 16*((kin>>3)&1), slot = (kin&7) + 8*(kin>>4)
__global__ void __launch_bounds__(256) k_xproj(const float* __restrict__ x,
                                               const float* __restrict__ bi,
                                               const unsigned short* __restrict__ wipk,
                                               unsigned short* __restrict__ xpk) {
  extern __shared__ char smem[];
  float*          xs  = (float*)smem;                        // 16 x 256 f32
  unsigned short* apk = (unsigned short*)(smem + 16384);     // 8 kt * 512
  unsigned short* opk = (unsigned short*)(smem + 24576);     // 16 kt * 512
  int blk = blockIdx.x;                // t*16 + g
  int t = blk >> 4, g = blk & 15;
  int tid = threadIdx.x;
  // stage the 16 batch rows of x at time t
  {
    int row = tid >> 4, k0 = (tid & 15) * 16;
    const float* src = x + ((size_t)(g * 16 + row) * T_ + t) * DIN_ + k0;
    float* dst = xs + row * DIN_ + k0;
    #pragma unroll
    for (int i = 0; i < 16; ++i) dst[i] = src[i];
  }
  __syncthreads();
  // pack A fragments for K=256
  for (int e = tid * 16, i = 0; i < 16; ++i, ++e) {
    int kt = e >> 9, lane = (e >> 4) & 31, slot = e & 15;
    int m = lane & 15;
    int kin = (slot & 7) | (((lane >> 4) & 1) << 3) | ((slot >> 3) << 4);
    apk[e] = f2bf(xs[m * DIN_ + kt * 32 + kin]);
  }
  __syncthreads();
  int wv = tid >> 5, lane = tid & 31;
  for (int nt = wv * 4; nt < wv * 4 + 4; ++nt) {
    float bv = bi[nt * 16 + (lane & 15)];
    v8f acc = {bv, bv, bv, bv, bv, bv, bv, bv};
    #pragma unroll
    for (int kt = 0; kt < 8; ++kt) {
      Frag a, b;
      const unsigned short* ap = apk + kt * 512 + lane * 16;
      a.q[0] = ((const uint4*)ap)[0]; a.q[1] = ((const uint4*)ap)[1];
      const unsigned short* bp = wipk + (size_t)(kt * 32 + nt) * 512 + lane * 16;
      b.q[0] = ((const uint4*)bp)[0]; b.q[1] = ((const uint4*)bp)[1];
      acc = __builtin_amdgcn_wmma_f32_16x16x32_bf16(false, a.bf, false, b.bf,
                                                    (short)0, acc, false, false);
    }
    // scatter result into packed A-fragment order (output cols become next K dim)
    #pragma unroll
    for (int j = 0; j < 8; ++j) {
      int M = j + ((lane >> 4) << 3);
      int col = nt * 16 + (lane & 15);
      int kin = col & 31, kt2 = col >> 5;
      int lane2 = M + 16 * ((kin >> 3) & 1);
      int slot2 = (kin & 7) + 8 * (kin >> 4);
      opk[kt2 * 512 + lane2 * 16 + slot2] = f2bf(acc[j]);
    }
  }
  __syncthreads();
  {
    const uint4* s4 = (const uint4*)opk;
    uint4* d4 = (uint4*)(xpk + (size_t)blk * 16 * 512);
    for (int i = tid; i < 1024; i += 256) d4[i] = s4[i];
  }
}

// ---------------- fused z+r GEMM: shared A fragments, two B streams ----------------
__device__ __forceinline__ void gemm_zr(const unsigned short* lowA,   // 16 kt (LDS)
                                        const unsigned short* upA,    // 16 kt (LDS)
                                        const unsigned short* B0,
                                        const unsigned short* B1,
                                        int ntg, int lane, v8f& acc0, v8f& acc1) {
  #pragma unroll 4
  for (int kt = 0; kt < 32; ++kt) {
    const unsigned short* ap =
        (kt < 16 ? lowA + kt * 512 : upA + (kt - 16) * 512) + lane * 16;
    Frag a, b0, b1;
    a.q[0] = ((const uint4*)ap)[0]; a.q[1] = ((const uint4*)ap)[1];
    const unsigned short* bp0 = B0 + (size_t)(kt * 32 + ntg) * 512 + lane * 16;
    const unsigned short* bp1 = B1 + (size_t)(kt * 32 + ntg) * 512 + lane * 16;
    b0.q[0] = ((const uint4*)bp0)[0]; b0.q[1] = ((const uint4*)bp0)[1];
    b1.q[0] = ((const uint4*)bp1)[0]; b1.q[1] = ((const uint4*)bp1)[1];
    acc0 = __builtin_amdgcn_wmma_f32_16x16x32_bf16(false, a.bf, false, b0.bf,
                                                   (short)0, acc0, false, false);
    acc1 = __builtin_amdgcn_wmma_f32_16x16x32_bf16(false, a.bf, false, b1.bf,
                                                   (short)0, acc1, false, false);
  }
}

__device__ __forceinline__ v8f gemm32(const unsigned short* lowA,
                                      const unsigned short* upA,
                                      const unsigned short* Bbase,
                                      int ntg, int lane, v8f acc) {
  #pragma unroll 4
  for (int kt = 0; kt < 32; ++kt) {
    const unsigned short* ap =
        (kt < 16 ? lowA + kt * 512 : upA + (kt - 16) * 512) + lane * 16;
    Frag a, b;
    a.q[0] = ((const uint4*)ap)[0]; a.q[1] = ((const uint4*)ap)[1];
    const unsigned short* bp = Bbase + (size_t)(kt * 32 + ntg) * 512 + lane * 16;
    b.q[0] = ((const uint4*)bp)[0]; b.q[1] = ((const uint4*)bp)[1];
    acc = __builtin_amdgcn_wmma_f32_16x16x32_bf16(false, a.bf, false, b.bf,
                                                  (short)0, acc, false, false);
  }
  return acc;
}

// ---------------- persistent recurrent scan ----------------
// grid: 64 WGs = 16 batch groups x 4 column parts; 256 threads = 8 waves
__global__ void __launch_bounds__(256) k_scan(const unsigned short* __restrict__ xpk,
                                              const unsigned short* __restrict__ wpk,
                                              const float* __restrict__ bz,
                                              const float* __restrict__ br,
                                              const float* __restrict__ bc,
                                              const float* __restrict__ gamma,
                                              const float* __restrict__ beta,
                                              float* __restrict__ out_hs,
                                              float* __restrict__ xrh,
                                              float* __restrict__ xh,
                                              float* __restrict__ xst,
                                              unsigned* __restrict__ cnt) {
  extern __shared__ char smem[];
  unsigned short* xstage = (unsigned short*)(smem + S_XSTAGE);
  unsigned short* hpk0   = (unsigned short*)(smem + S_HPK0);
  unsigned short* hpk1   = (unsigned short*)(smem + S_HPK1);
  unsigned short* rhpk   = (unsigned short*)(smem + S_RHPK);
  unsigned short* hstate = (unsigned short*)(smem + S_HSTATE);
  unsigned short* hnb    = (unsigned short*)(smem + S_HNB);
  float* gb   = (float*)(smem + S_GB);
  float* psum = (float*)(smem + S_PSUM);
  float* psq  = (float*)(smem + S_PSQ);
  float* mu   = (float*)(smem + S_MU);
  float* rstd = (float*)(smem + S_RSTD);

  int g    = blockIdx.x >> 2;
  int part = blockIdx.x & 3;
  int tid  = threadIdx.x;
  int wv = tid >> 5, lane = tid & 31;
  int b0 = g * 16;
  unsigned* mycnt = cnt + g;
  unsigned phase = 0;

  int ntg  = part * 8 + wv;              // global n-tile 0..31 for this wave
  int colg = ntg * 16 + (lane & 15);     // global output column
  int cll  = wv * 16 + (lane & 15);      // local column within the 128-col part

  // init: h = 0, gamma/beta table
  for (int i = tid; i < 16384; i += 256) hpk0[i] = 0;  // hpk0 + hpk1 (contiguous)
  for (int i = tid; i < 4096; i += 256)  hstate[i] = 0;
  for (int i = tid; i < 2048; i += 256) {
    int l = i >> 10, r = i & 1023;
    gb[i] = (r < 512) ? gamma[l * H_ + r] : beta[l * H_ + (r - 512)];
  }
  // hoist per-column biases into registers
  float bzv[L_], brv[L_], bcv[L_];
  #pragma unroll
  for (int l = 0; l < L_; ++l) {
    bzv[l] = bz[l * H_ + colg];
    brv[l] = br[l * H_ + colg];
    bcv[l] = bc[l * H_ + colg];
  }
  // prologue: async prefetch of the packed x tile for t=0 into buffer 0
  for (int i = tid; i < 1024; i += 256)
    async_ld_b128(xpk + (size_t)(0 * 16 + g) * 8192 + i * 8,
                  xstage + i * 8, (unsigned)(S_XSTAGE + i * 16));
  __syncthreads();

  for (int t = 0; t < T_; ++t) {
    // retire async loads for this step's x tile, then prefetch t+1
    wait_asynccnt0();
    if (t + 1 < T_) {
      const unsigned short* xnext = xpk + (size_t)((t + 1) * 16 + g) * 8192;
      int nb = ((t + 1) & 1) * 8192;
      for (int i = tid; i < 1024; i += 256)
        async_ld_b128(xnext + i * 8, xstage + nb + i * 8,
                      (unsigned)(S_XSTAGE + nb * 2 + i * 16));
    }
    __syncthreads();
    const unsigned short* xlow = xstage + (t & 1) * 8192;

    for (int l = 0; l < L_; ++l) {
      const unsigned short* lowA = (l == 0) ? xlow : hpk0;   // comb lower half: xt
      unsigned short* upA  = (l == 0) ? hpk0 : hpk1;         // comb upper half: h_prev(l)
      const unsigned short* wzb = wpk + (size_t)(l * 3 + 0) * (32 * 32 * 512);
      const unsigned short* wrb = wpk + (size_t)(l * 3 + 1) * (32 * 32 * 512);
      const unsigned short* wcb = wpk + (size_t)(l * 3 + 2) * (32 * 32 * 512);
      unsigned short* hs_l = hstate + l * 2048;              // [16][128] bf16

      // ---- z and r gates (fused, shared A fragments) ----
      v8f zac = {bzv[l], bzv[l], bzv[l], bzv[l], bzv[l], bzv[l], bzv[l], bzv[l]};
      v8f rac = {brv[l], brv[l], brv[l], brv[l], brv[l], brv[l], brv[l], brv[l]};
      gemm_zr(lowA, upA, wzb, wrb, ntg, lane, zac, rac);

      v8f z;
      #pragma unroll
      for (int j = 0; j < 8; ++j) {
        int M = j + ((lane >> 4) << 3);
        float zz = 1.f / (1.f + __expf(-zac[j]));
        z[j] = zz;
        float rr = 1.f / (1.f + __expf(-rac[j]));
        float hv = bf2f(hs_l[M * 128 + cll]);
        xrh[(((size_t)(g * 4 + part)) * 16 + M) * 128 + cll] = rr * hv;
      }
      group_barrier(mycnt, 4u * (++phase));

      // ---- gather all 512 cols of r*h, pack into A-fragment order ----
      for (int e = tid; e < 8192; e += 256) {
        int row = e >> 9, kcol = e & 511;
        float v = xrh[(((size_t)(g * 4 + (kcol >> 7))) * 16 + row) * 128 + (kcol & 127)];
        int kin = kcol & 31, kt2 = kcol >> 5;
        int lane2 = row + 16 * ((kin >> 3) & 1);
        int slot2 = (kin & 7) + 8 * (kin >> 4);
        rhpk[kt2 * 512 + lane2 * 16 + slot2] = f2bf(v);
      }
      __syncthreads();

      // ---- candidate and h_new ----
      v8f cac = {bcv[l], bcv[l], bcv[l], bcv[l], bcv[l], bcv[l], bcv[l], bcv[l]};
      cac = gemm32(lowA, rhpk, wcb, ntg, lane, cac);

      #pragma unroll
      for (int j = 0; j < 8; ++j) {
        int M = j + ((lane >> 4) << 3);
        float ht = tanhf(cac[j]);
        float hv = bf2f(hs_l[M * 128 + cll]);
        float hnv = (1.f - z[j]) * hv + z[j] * ht;
        hnb[M * 128 + cll] = f2bf(hnv);
      }
      __syncthreads();

      // ---- layernorm partial stats over local 128 cols + publish h_new ----
      {
        int row = tid & 15, ch = tid >> 4;
        float s = 0.f, q = 0.f;
        #pragma unroll
        for (int i = 0; i < 8; ++i) {
          float v = bf2f(hnb[row * 128 + ch * 8 + i]);
          s += v; q += v * v;
        }
        psum[row * 16 + ch] = s; psq[row * 16 + ch] = q;
      }
      for (int e = tid; e < 2048; e += 256)
        xh[((size_t)(g * 4 + part)) * 2048 + e] = bf2f(hnb[e]);
      __syncthreads();
      if (tid < 16) {
        float s = 0.f, q = 0.f;
        for (int ch = 0; ch < 16; ++ch) { s += psum[tid * 16 + ch]; q += psq[tid * 16 + ch]; }
        size_t o = (((size_t)(g * 4 + part)) * 16 + tid) * 2;
        xst[o] = s; xst[o + 1] = q;
      }
      group_barrier(mycnt, 4u * (++phase));

      // ---- combine stats across parts ----
      if (tid < 16) {
        float s = 0.f, q = 0.f;
        for (int p = 0; p < 4; ++p) {
          size_t o = (((size_t)(g * 4 + p)) * 16 + tid) * 2;
          s += xst[o]; q += xst[o + 1];
        }
        float m = s / (float)H_;
        float var = q / (float)H_ - m * m;
        mu[tid] = m;
        rstd[tid] = rsqrtf(var + EPS_);
      }
      __syncthreads();

      // ---- normalized h (all 512 cols) -> packed A-fragments for the next GEMM ----
      unsigned short* hdst = (l == 0) ? hpk0 : hpk1;
      for (int e = tid; e < 8192; e += 256) {
        int row = e >> 9, kcol = e & 511;
        float v = xh[((size_t)(g * 4 + (kcol >> 7))) * 2048 + row * 128 + (kcol & 127)];
        float nv = (v - mu[row]) * rstd[row] * gb[l * 1024 + kcol] + gb[l * 1024 + 512 + kcol];
        int kin = kcol & 31, kt2 = kcol >> 5;
        int lane2 = row + 16 * ((kin >> 3) & 1);
        int slot2 = (kin & 7) + 8 * (kin >> 4);
        hdst[kt2 * 512 + lane2 * 16 + slot2] = f2bf(nv);
      }
      // ---- local normalized state + hidden_states output (layer 1 only) ----
      for (int e = tid; e < 2048; e += 256) {
        int row = e >> 7, c = e & 127;
        int kcol = part * 128 + c;
        float v = bf2f(hnb[row * 128 + c]);
        float nv = (v - mu[row]) * rstd[row] * gb[l * 1024 + kcol] + gb[l * 1024 + 512 + kcol];
        hs_l[row * 128 + c] = f2bf(nv);
        if (l == 1)
          out_hs[((size_t)(b0 + row) * T_ + t) * H_ + kcol] = nv;
      }
      __syncthreads();
    }
  }
}

// ---------------- heads: beliefs, actions, final_hidden ----------------
__global__ void k_head(const float* __restrict__ hs, const float* __restrict__ Wb,
                       const float* __restrict__ bb, const float* __restrict__ Wa,
                       const float* __restrict__ ba, float* __restrict__ beliefs,
                       float* __restrict__ actions, float* __restrict__ fh) {
  __shared__ float last[H_];
  __shared__ float red[DOUT_];
  int b = blockIdx.x, j = threadIdx.x;  // 64 threads
  const float* src = hs + ((size_t)b * T_ + (T_ - 1)) * H_;
  for (int i = j; i < H_; i += DOUT_) last[i] = src[i];
  __syncthreads();
  float acc = bb[j];
  for (int c = 0; c < H_; ++c) acc += last[c] * Wb[c * DOUT_ + j];
  beliefs[b * DOUT_ + j] = 1.f / (1.f + __expf(-acc));
  float pa = 0.f;
  for (int c = j; c < H_; c += DOUT_) pa += last[c] * Wa[c];
  red[j] = pa;
  __syncthreads();
  if (j == 0) {
    float s = ba[0];
    for (int i = 0; i < DOUT_; ++i) s += red[i];
    actions[b] = 1.f / (1.f + __expf(-s));
  }
  for (int i = j; i < H_; i += DOUT_) fh[b * H_ + i] = last[i];
}

extern "C" void kernel_launch(void* const* d_in, const int* in_sizes, int n_in,
                              void* d_out, int out_size, void* d_ws, size_t ws_size,
                              hipStream_t stream) {
  (void)in_sizes; (void)n_in; (void)out_size; (void)ws_size;
  const float* x     = (const float*)d_in[0];
  const float* Wi    = (const float*)d_in[1];
  const float* bi    = (const float*)d_in[2];
  const float* Wz    = (const float*)d_in[3];
  const float* bz    = (const float*)d_in[4];
  const float* Wr    = (const float*)d_in[5];
  const float* br    = (const float*)d_in[6];
  const float* Wc    = (const float*)d_in[7];
  const float* bc    = (const float*)d_in[8];
  const float* gamma = (const float*)d_in[9];
  const float* beta  = (const float*)d_in[10];
  const float* Wb    = (const float*)d_in[11];
  const float* bb    = (const float*)d_in[12];
  const float* Wa    = (const float*)d_in[13];
  const float* ba    = (const float*)d_in[14];

  char* ws = (char*)d_ws;
  unsigned short* xpk  = (unsigned short*)(ws + XPK_OFF);
  unsigned short* wpk  = (unsigned short*)(ws + WPK_OFF);
  unsigned short* wipk = (unsigned short*)(ws + WIPK_OFF);
  float* xrh = (float*)(ws + XRH_OFF);
  float* xh  = (float*)(ws + XH_OFF);
  float* xst = (float*)(ws + XST_OFF);
  unsigned* cnt = (unsigned*)(ws + CNT_OFF);

  float* out     = (float*)d_out;
  float* out_hs  = out;
  float* out_bel = out + (size_t)B_ * T_ * H_;
  float* out_act = out_bel + (size_t)B_ * DOUT_;
  float* out_fh  = out_act + B_;

  k_zero<<<1, 64, 0, stream>>>(cnt);
  k_pack_w<<<(2 * 3 * 32 * 32 * 512) / 256, 256, 0, stream>>>(Wz, Wr, Wc, wpk);
  k_pack_wi<<<(8 * 32 * 512) / 256, 256, 0, stream>>>(Wi, wipk);
  k_xproj<<<T_ * GROUPS_, 256, 40960, stream>>>(x, bi, wipk, xpk);
  k_scan<<<GROUPS_ * PARTS_, 256, S_TOTAL, stream>>>(xpk, wpk, bz, br, bc, gamma, beta,
                                                     out_hs, xrh, xh, xst, cnt);
  k_head<<<B_, DOUT_, 0, stream>>>(out_hs, Wb, bb, Wa, ba, out_bel, out_act, out_fh);
}
